// MultiLevelMamba_88021059764655
// MI455X (gfx1250) — matrix operand order repaired and checked by reading, not compiled, not run on hardware
//
#include <hip/hip_runtime.h>
#include <math.h>

// Problem dims (fixed by the reference)
#define BD   8
#define ND   4096
#define LD   32        // lookback (time)
#define DD   64        // d_in
#define SD   64        // d_state
#define NLV  2         // levels
#define TOTAL_BN (BD*ND)        // 32768

#define STRIDE 72               // padded half-stride (144B: 16B aligned, conflict-free)
#define WAVES_PER_WG 4
#define THREADS (WAVES_PER_WG*32)
#define NUM_WG 256
#define NUM_WAVES (NUM_WG*WAVES_PER_WG)

// ---- LDS layout (bytes) ----
#define LPWT_OFF 0
#define LPWT_SZ  (NLV*DD*STRIDE*2)        // 18432  lpW^T f16 [lvl][e][d]
#define PWT_OFF  (LPWT_OFF + LPWT_SZ)
#define PWT_SZ   (NLV*3*SD*STRIDE*2)      // 55296  pW^T f16 [lvl][col][d]
#define OW_OFF   (PWT_OFF + PWT_SZ)
#define OW_SZ    (NLV*SD*SD*2)            // 16384  oW f16 [lvl][k][j]
#define RW_OFF   (OW_OFF + OW_SZ)
#define RW_SZ    (NLV*SD*SD*2)            // 16384
#define WAVE_OFF (RW_OFF + RW_SZ)         // 106496
// per-wave scratch
#define XT_OFF   0
#define XT_SZ    (LD*STRIDE*2)            // 4608  x tile f16 [t][d] row-major
#define AT_OFF   (XT_OFF + XT_SZ)
#define BT_OFF   (AT_OFF + XT_SZ)
#define XM_OFF   (BT_OFF + XT_SZ)         // 13824  xm f32[32]
#define CL_OFF   (XM_OFF + LD*4)          // 13952  c_last f32[64]
#define WAVE_SZ  (CL_OFF + SD*4)          // 14208
#define SMEM_BYTES (WAVE_OFF + WAVES_PER_WG*WAVE_SZ)   // 163328 (2 WGs / 320KB WGP)

typedef __attribute__((ext_vector_type(16))) _Float16 v16h;
typedef __attribute__((ext_vector_type(8)))  _Float16 v8h;
typedef __attribute__((ext_vector_type(8)))  float    v8f;
typedef __attribute__((ext_vector_type(4)))  float    v4f;

__device__ __forceinline__ void lds_fence() {
  asm volatile("s_wait_dscnt 0" ::: "memory");
}

__device__ __forceinline__ v16h comb16(v8h lo, v8h hi8) {
  v16h r;
#pragma unroll
  for (int e = 0; e < 8; ++e) { r[e] = lo[e]; r[e + 8] = hi8[e]; }
  return r;
}

// A fragment (16x32 f16) from LDS weight row base: K = kc*32 + hi*16 + e, 16 contiguous halves
__device__ __forceinline__ v16h afrag_lds(const _Float16* wrow, int kc, int kh) {
  v8h c0 = *(const v8h*)(wrow + kc * 32 + kh * 16);
  v8h c1 = *(const v8h*)(wrow + kc * 32 + kh * 16 + 8);
  return comb16(c0, c1);
}

// B fragment (32x16 f16) from LDS row (x[t][.]): K = kc*32 + hi*16 + e, 16 contiguous halves
__device__ __forceinline__ v16h bfrag_lds(const _Float16* xrow, int kc, int kb) {
  v8h c0 = *(const v8h*)(xrow + kc * 32 + kb * 16);
  v8h c1 = *(const v8h*)(xrow + kc * 32 + kb * 16 + 8);
  return comb16(c0, c1);
}

// B fragment from global f32 row chunk (16 contiguous floats -> f16)
__device__ __forceinline__ v16h bfrag_global(const float* base) {
  v4f A = *(const v4f*)base,       B = *(const v4f*)(base + 4);
  v4f C = *(const v4f*)(base + 8), D = *(const v4f*)(base + 12);
  v16h f;
#pragma unroll
  for (int e = 0; e < 4; ++e) {
    f[e] = (_Float16)A[e];      f[e + 4]  = (_Float16)B[e];
    f[e + 8] = (_Float16)C[e];  f[e + 12] = (_Float16)D[e];
  }
  return f;
}

__device__ __forceinline__ v8f wmma32(v16h a, v16h b, v8f c) {
  return __builtin_amdgcn_wmma_f32_16x16x32_f16(false, a, false, b, (short)0, c, false, false);
}

// pack 8 accumulator lanes + bias into 8 contiguous halves
__device__ __forceinline__ v8h pack8(const v8f acc, v4f b0, v4f b1) {
  v8h r;
#pragma unroll
  for (int e = 0; e < 4; ++e) {
    r[e]     = (_Float16)(acc[e]     + b0[e]);
    r[e + 4] = (_Float16)(acc[e + 4] + b1[e]);
  }
  return r;
}

__device__ __forceinline__ v8h pack8_sigmoid(const v8f acc, v4f b0, v4f b1) {
  v8h r;
#pragma unroll
  for (int e = 0; e < 4; ++e) {
    r[e]     = (_Float16)(1.0f / (1.0f + __expf(-(acc[e]     + b0[e]))));
    r[e + 4] = (_Float16)(1.0f / (1.0f + __expf(-(acc[e + 4] + b1[e]))));
  }
  return r;
}

__device__ __forceinline__ float gelu_exact(float v) {
  return 0.5f * v * (1.0f + erff(v * 0.70710678118654752f));
}

__global__ void __launch_bounds__(THREADS)
MultiLevelMamba_88021059764655_kernel(
    const float* __restrict__ p_seq, const float* __restrict__ lp_W,
    const float* __restrict__ lp_b,  const float* __restrict__ pW,
    const float* __restrict__ pb,    const float* __restrict__ oW,
    const float* __restrict__ ob,    const float* __restrict__ rW,
    const float* __restrict__ rb,    const float* __restrict__ fW,
    const float* __restrict__ fb,    float* __restrict__ out) {
  extern __shared__ char smem[];
  _Float16* lpWT = (_Float16*)(smem + LPWT_OFF);
  _Float16* pWT  = (_Float16*)(smem + PWT_OFF);
  _Float16* oW16 = (_Float16*)(smem + OW_OFF);
  _Float16* rW16 = (_Float16*)(smem + RW_OFF);
  char* wbase = smem + WAVE_OFF + (size_t)(threadIdx.x >> 5) * WAVE_SZ;
  _Float16* x_lds = (_Float16*)(wbase + XT_OFF);   // x[t][d] row-major
  _Float16* a_lds = (_Float16*)(wbase + AT_OFF);   // a[t][s] row-major (post-sigmoid)
  _Float16* b_lds = (_Float16*)(wbase + BT_OFF);   // b[t][s]
  float* xm_lds = (float*)(wbase + XM_OFF);
  float* c_lds  = (float*)(wbase + CL_OFF);
  float* vec    = (float*)x_lds;  // tail scratch reuses x tile (x dead by then)

  // ---- cooperative weight staging (once per workgroup, both levels) ----
  const int tid = threadIdx.x;
  for (int i = tid; i < NLV * DD * DD; i += THREADS) {   // lpW [lvl][d][e] -> lpWT[lvl][e][d]
    int lvl = i >> 12, rem = i & 4095, d = rem >> 6, e = rem & 63;
    lpWT[(lvl * DD + e) * STRIDE + d] = (_Float16)lp_W[i];
  }
  for (int i = tid; i < NLV * DD * 3 * SD; i += THREADS) { // pW [lvl][d][c] -> pWT[lvl][c][d]
    int lvl = i / (DD * 3 * SD), rem = i % (DD * 3 * SD);
    int d = rem / (3 * SD), c = rem % (3 * SD);
    pWT[(lvl * 3 * SD + c) * STRIDE + d] = (_Float16)pW[i];
  }
  for (int i = tid; i < NLV * SD * SD; i += THREADS) {
    oW16[i] = (_Float16)oW[i];
    rW16[i] = (_Float16)rW[i];
  }
  __syncthreads();

  const int lane = threadIdx.x & 31;
  const int nlo  = lane & 15;   // within-tile N (= t index for our transposed GEMMs)
  const int hi   = lane >> 4;   // half-wave selector
  const int waveGid = blockIdx.x * WAVES_PER_WG + (threadIdx.x >> 5);
  const float fbv = fb[0];

  for (int bn = waveGid; bn < TOTAL_BN; bn += NUM_WAVES) {
    const float* Pbase = p_seq + (size_t)bn * (LD * DD);
    if (bn + NUM_WAVES < TOTAL_BN)   // global_prefetch_b8 for next tile
      __builtin_prefetch(p_seq + (size_t)(bn + NUM_WAVES) * (LD * DD) + lane * 64, 0, 1);

    // ---- P^T B-fragments: lane reads 16 contiguous floats of row t, once per bn ----
    // nt: t-tile (t = nt*16 + nlo); kc: K(d)-chunk; K = kc*32 + hi*16 + e
    v16h pfrag[2][2];
#pragma unroll
    for (int nt = 0; nt < 2; ++nt) {
      const float* row = Pbase + (size_t)(nt * 16 + nlo) * DD;
#pragma unroll
      for (int kc = 0; kc < 2; ++kc)
        pfrag[nt][kc] = bfrag_global(row + kc * 32 + hi * 16);
    }

    float y = fbv;
    for (int level = 0; level < NLV; ++level) {
      // ========== GEMM1': x^T = lpW^T @ P^T  ->  x[t][d] row-major in LDS ==========
#pragma unroll
      for (int mt = 0; mt < 4; ++mt) {   // e-tiles (output d dimension)
        const _Float16* wrow = lpWT + (size_t)(level * DD + mt * 16 + nlo) * STRIDE;
        v16h a0 = afrag_lds(wrow, 0, hi);
        v16h a1 = afrag_lds(wrow, 1, hi);
        const float* bp = lp_b + level * DD + mt * 16 + hi * 8;  // 8 consecutive e per lane
        v4f bb0 = *(const v4f*)bp, bb1 = *(const v4f*)(bp + 4);
#pragma unroll
        for (int nt = 0; nt < 2; ++nt) {
          v8f acc = {0.f, 0.f, 0.f, 0.f, 0.f, 0.f, 0.f, 0.f};
          acc = wmma32(a0, pfrag[nt][0], acc);
          acc = wmma32(a1, pfrag[nt][1], acc);
          // D: lane holds e = mt*16 + hi*8 + r (contiguous), t = nt*16 + nlo
          *(v8h*)(x_lds + (size_t)(nt * 16 + nlo) * STRIDE + mt * 16 + hi * 8) =
              pack8(acc, bb0, bb1);
        }
      }
      lds_fence();

      // xm[t] = mean_d x[t,d]  (lane t, contiguous v8h reads)
      {
        float s = 0.f;
#pragma unroll
        for (int j = 0; j < 8; ++j) {
          v8h v = *(const v8h*)(x_lds + (size_t)lane * STRIDE + j * 8);
#pragma unroll
          for (int e = 0; e < 8; ++e) s += (float)v[e];
        }
        xm_lds[lane] = s * (1.0f / DD);
      }
      lds_fence();

      // ========== GEMM2': p^T = pW^T @ x^T  ->  a[t][s], b[t][s], c_last[s] ==========
      // x B-fragments: 16 contiguous halves of x[t] rows, hoisted (reused by all 12 M-tiles)
      v16h xb[2][2];
#pragma unroll
      for (int nt = 0; nt < 2; ++nt)
#pragma unroll
        for (int kc = 0; kc < 2; ++kc)
          xb[nt][kc] = bfrag_lds(x_lds + (size_t)(nt * 16 + nlo) * STRIDE, kc, hi);

#pragma unroll
      for (int mt = 0; mt < 12; ++mt) {  // output-column tiles: 0-3 a, 4-7 b, 8-11 c
        const _Float16* wrow = pWT + (size_t)(level * 3 * SD + mt * 16 + nlo) * STRIDE;
        v16h a0 = afrag_lds(wrow, 0, hi);
        v16h a1 = afrag_lds(wrow, 1, hi);
        const float* bp = pb + level * 3 * SD + mt * 16 + hi * 8;
        v4f bb0 = *(const v4f*)bp, bb1 = *(const v4f*)(bp + 4);
        if (mt < 4) {            // part a -> sigmoid -> a_lds
#pragma unroll
          for (int nt = 0; nt < 2; ++nt) {
            v8f acc = {0.f, 0.f, 0.f, 0.f, 0.f, 0.f, 0.f, 0.f};
            acc = wmma32(a0, xb[nt][0], acc);
            acc = wmma32(a1, xb[nt][1], acc);
            *(v8h*)(a_lds + (size_t)(nt * 16 + nlo) * STRIDE + mt * 16 + hi * 8) =
                pack8_sigmoid(acc, bb0, bb1);
          }
        } else if (mt < 8) {     // part b -> b_lds  (s = col - 64)
#pragma unroll
          for (int nt = 0; nt < 2; ++nt) {
            v8f acc = {0.f, 0.f, 0.f, 0.f, 0.f, 0.f, 0.f, 0.f};
            acc = wmma32(a0, xb[nt][0], acc);
            acc = wmma32(a1, xb[nt][1], acc);
            *(v8h*)(b_lds + (size_t)(nt * 16 + nlo) * STRIDE + (mt - 4) * 16 + hi * 8) =
                pack8(acc, bb0, bb1);
          }
        } else {                 // part c: only t=31 (nt=1, lanes nlo==15)
          v8f acc = {0.f, 0.f, 0.f, 0.f, 0.f, 0.f, 0.f, 0.f};
          acc = wmma32(a0, xb[1][0], acc);
          acc = wmma32(a1, xb[1][1], acc);
          if (nlo == 15) {       // s = col - 128 = (mt-8)*16 + hi*8 + r
            float* cp = c_lds + (mt - 8) * 16 + hi * 8;
            v4f c0 = {acc[0] + bb0[0], acc[1] + bb0[1], acc[2] + bb0[2], acc[3] + bb0[3]};
            v4f c1 = {acc[4] + bb1[0], acc[5] + bb1[1], acc[6] + bb1[2], acc[7] + bb1[3]};
            *(v4f*)cp = c0;
            *(v4f*)(cp + 4) = c1;
          }
        }
      }
      lds_fence();

      // ========== selective scan over t (2 states per lane) ==========
      float h1 = 0.f, h2 = 0.f;
      for (int t = 0; t < LD; ++t) {
        float xv = xm_lds[t];
        h1 = (float)a_lds[t * STRIDE + lane]      * h1 + (float)b_lds[t * STRIDE + lane]      * xv;
        h2 = (float)a_lds[t * STRIDE + lane + 32] * h2 + (float)b_lds[t * STRIDE + lane + 32] * xv;
      }

      // ========== tail: gelu(c*h) @ oW + ob, + mean, @ rW + rb ==========
      float g1 = gelu_exact(c_lds[lane] * h1);
      float g2 = gelu_exact(c_lds[lane + 32] * h2);
      vec[lane] = g1; vec[lane + 32] = g2;
      lds_fence();
      float o1 = ob[level * SD + lane], o2 = ob[level * SD + lane + 32];
#pragma unroll 8
      for (int k = 0; k < SD; ++k) {
        float gk = vec[k];
        o1 += gk * (float)oW16[(level * SD + k) * SD + lane];
        o2 += gk * (float)oW16[(level * SD + k) * SD + lane + 32];
      }
      float xl = xm_lds[LD - 1];
      float l1 = o1 + xl, l2 = o2 + xl;
      lds_fence();                 // all lanes done reading vec (lockstep) before overwrite
      vec[lane] = l1; vec[lane + 32] = l2;
      lds_fence();
      float f1 = rb[level * SD + lane], f2 = rb[level * SD + lane + 32];
#pragma unroll 8
      for (int k = 0; k < SD; ++k) {
        float lk = vec[k];
        f1 += lk * (float)rW16[(level * SD + k) * SD + lane];
        f2 += lk * (float)rW16[(level * SD + k) * SD + lane + 32];
      }
      float pd = f1 * fW[level * SD + lane] + f2 * fW[level * SD + lane + 32];
#pragma unroll
      for (int off = 16; off > 0; off >>= 1) pd += __shfl_xor(pd, off, 32);
      y += pd;
    }
    if (lane == 0) out[bn] = y;
  }
}

extern "C" void kernel_launch(void* const* d_in, const int* in_sizes, int n_in,
                              void* d_out, int out_size, void* d_ws, size_t ws_size,
                              hipStream_t stream) {
  (void)in_sizes; (void)n_in; (void)d_ws; (void)ws_size; (void)out_size;
  const float* p_seq = (const float*)d_in[0];
  const float* lp_W  = (const float*)d_in[1];
  const float* lp_b  = (const float*)d_in[2];
  const float* pW    = (const float*)d_in[3];
  const float* pb    = (const float*)d_in[4];
  const float* oW    = (const float*)d_in[5];
  const float* ob    = (const float*)d_in[6];
  const float* rW    = (const float*)d_in[7];
  const float* rb    = (const float*)d_in[8];
  const float* fW    = (const float*)d_in[9];
  const float* fb    = (const float*)d_in[10];
  float* out = (float*)d_out;

  (void)hipFuncSetAttribute((const void*)MultiLevelMamba_88021059764655_kernel,
                            hipFuncAttributeMaxDynamicSharedMemorySize, SMEM_BYTES);
  MultiLevelMamba_88021059764655_kernel<<<NUM_WG, THREADS, SMEM_BYTES, stream>>>(
      p_seq, lp_W, lp_b, pW, pb, oW, ob, rW, rb, fW, fb, out);
}
